// ScaledDotProductAttention_87522843558380
// MI455X (gfx1250) — matrix operand order repaired and checked by reading, not compiled, not run on hardware
//
#include <hip/hip_runtime.h>
#include <hip/hip_bf16.h>
#include <stdint.h>

typedef __attribute__((ext_vector_type(16))) __bf16 v16bf;
typedef __attribute__((ext_vector_type(8)))  __bf16 v8bf;
typedef __attribute__((ext_vector_type(16))) float  v16f;
typedef __attribute__((ext_vector_type(8)))  float  v8f;
typedef __attribute__((ext_vector_type(4)))  float  v4f;
typedef __attribute__((ext_vector_type(4)))  unsigned int u32x4;
typedef __attribute__((ext_vector_type(8)))  int    i32x8;
typedef __attribute__((ext_vector_type(4)))  int    i32x4;

#define S_LEN   4096
#define DK      64
#define NWAVES  8
#define QTILE   32
#define PSTRIDE 4104          // 4096 + 8 bf16 pad -> conflict-free strided LDS access
#define QSTRIDE 65            // f32 Q staging stride (TDM pad: 1 dword per 64)

#define PS_OFF    0
#define PS_BYTES  (QTILE * PSTRIDE * 2)            // 262656
#define QSF_OFF   (PS_OFF + PS_BYTES)
#define QSF_BYTES (QTILE * QSTRIDE * 4)            // 8320
#define RED_OFF   (QSF_OFF + QSF_BYTES)
#define RED_BYTES (NWAVES * QTILE * 4)             // 1024
#define RMAX_OFF  (RED_OFF + RED_BYTES)
#define RINV_OFF  (RMAX_OFF + QTILE * 4)
#define OACC_OFF  (RINV_OFF + QTILE * 4)
#define OACC_N    (8 * 8 * 32)                     // [tile(2x4)][vgpr(8)][lane(32)]
#define LDS_BYTES (OACC_OFF + OACC_N * 4)          // 280448 < 320KB WGP LDS

union FragBf { v16bf v; v8bf h[2]; __bf16 e[16]; };

__global__ __launch_bounds__(256, 1)
void attn_fused_kernel(const float* __restrict__ Q, const float* __restrict__ K,
                       const float* __restrict__ V, float* __restrict__ out,
                       float* __restrict__ attn) {
  extern __shared__ __align__(16) char smem[];
  __bf16* Ps       = (__bf16*)(smem + PS_OFF);
  float*  Qs       = (float*)(smem + QSF_OFF);
  float*  redbuf   = (float*)(smem + RED_OFF);
  float*  rowmax_s = (float*)(smem + RMAX_OFF);
  float*  rowinv_s = (float*)(smem + RINV_OFF);
  float*  Oacc     = (float*)(smem + OACC_OFF);

  const int tid  = threadIdx.x;
  const int wave = tid >> 5;
  const int ln   = tid & 31;
  const int nn   = ln & 15;     // A-frag row / B,C-frag column within 16
  const int hh   = ln >> 4;     // lane half

  const int q0      = blockIdx.x * QTILE;
  const int bh      = blockIdx.y;
  const int nchunks = blockIdx.x + 1;       // causal: keys [0, q0+32) in 32-key chunks

  const float* Qbh = Q + (size_t)bh * S_LEN * DK;
  const float* Kbh = K + (size_t)bh * S_LEN * DK;
  const float* Vbh = V + (size_t)bh * S_LEN * DK;

  // ---- stage raw f32 Q tile into LDS (TDM DMA when available) ----
#if __has_builtin(__builtin_amdgcn_tensor_load_to_lds) && __has_builtin(__builtin_amdgcn_s_wait_tensorcnt)
  if (wave == 0) {
    const uint64_t ga = (uint64_t)(uintptr_t)(Qbh + (size_t)q0 * DK);
    const uint32_t lds_addr = (uint32_t)(uintptr_t)Qs;   // aperture low bits = LDS offset
    u32x4 g0;
    g0[0] = 1u;                                  // count=1, user descriptor
    g0[1] = lds_addr;
    g0[2] = (uint32_t)ga;
    g0[3] = (uint32_t)((ga >> 32) & 0x01FFFFFFu) | 0x80000000u;  // type=2 ("image")
    i32x8 g1;
    g1[0] = (int)((2u << 16) | (1u << 20) | (5u << 22)); // data_size=4B, pad_en, 64dw interval
    g1[1] = (int)(64u << 16);                    // tensor_dim0 = 64
    g1[2] = (int)(32u << 16);                    // tensor_dim1 = 32
    g1[3] = (int)(64u << 16);                    // tile_dim0 = 64
    g1[4] = 32;                                  // tile_dim1 = 32
    g1[5] = 64;                                  // tensor_dim0_stride = 64
    g1[6] = 0; g1[7] = 0;
    i32x4 z4; z4[0] = 0; z4[1] = 0; z4[2] = 0; z4[3] = 0;
#if __has_include(<hip/amd_detail/amd_gfx1250_TDM.h>)
    i32x8 z8; for (int j = 0; j < 8; ++j) z8[j] = 0;
    __builtin_amdgcn_tensor_load_to_lds(g0, g1, z4, z4, z8, 0);
#else
    __builtin_amdgcn_tensor_load_to_lds(g0, g1, z4, z4, 0);
#endif
    __builtin_amdgcn_s_wait_tensorcnt(0);
  }
#else
  for (int idx = tid; idx < QTILE * DK; idx += 256) {
    int r = idx >> 6, c = idx & 63;
    Qs[r * QSTRIDE + c] = Qbh[(size_t)(q0 + r) * DK + c];
  }
#endif
  __syncthreads();

  // ---- per-wave persistent Q A-fragments, scale 1/sqrt(64) folded in ----
  const float sc = 0.125f;
  FragBf aq[2][2];
  for (int mt = 0; mt < 2; ++mt)
    for (int dc = 0; dc < 2; ++dc) {
      const float* qp = &Qs[(mt * 16 + nn) * QSTRIDE + dc * 32 + 8 * hh];
      v16f qq;
      for (int j = 0; j < 8; ++j) qq[j]     = qp[j]      * sc;  // K = dc*32 + 8h + 0..7
      for (int j = 0; j < 8; ++j) qq[8 + j] = qp[16 + j] * sc;  // K = dc*32+16+8h + 0..7
      aq[mt][dc].v = __builtin_convertvector(qq, v16bf);
    }

  // ============ Phase A: S = (Q/sqrt(d))K^T (masked) -> LDS, row-max ============
  float rmax[2][8];
  for (int mt = 0; mt < 2; ++mt)
    for (int r = 0; r < 8; ++r) rmax[mt][r] = -3.0e38f;

#pragma unroll 2
  for (int c = wave; c < nchunks; c += NWAVES) {
    const int k0 = c * 32;
    if (c + NWAVES < nchunks)
      __builtin_prefetch(&Kbh[(size_t)(k0 + NWAVES * 32) * DK], 0, 0);

    FragBf bk[2][2];   // B-frag: lane holds key column nn, rows d = dc*32 + 8h + j (j<8), dc*32 + 16 + 8h + (j-8) (j>=8)
    for (int nt = 0; nt < 2; ++nt) {
      const size_t kb = (size_t)(k0 + nt * 16 + nn) * DK;
      for (int dc = 0; dc < 2; ++dc) {
        const v4f* kp = (const v4f*)&Kbh[kb + dc * 32 + 8 * hh];
        v4f f0 = kp[0], f1 = kp[1], f2 = kp[4], f3 = kp[5];
        v16f kk;
        for (int j = 0; j < 4; ++j) kk[j]      = f0[j];
        for (int j = 0; j < 4; ++j) kk[4 + j]  = f1[j];
        for (int j = 0; j < 4; ++j) kk[8 + j]  = f2[j];
        for (int j = 0; j < 4; ++j) kk[12 + j] = f3[j];
        bk[nt][dc].v = __builtin_convertvector(kk, v16bf);
      }
    }

    const bool isdiag = (c == nchunks - 1);   // only the diagonal chunk needs masking
    for (int mt = 0; mt < 2; ++mt)
      for (int nt = 0; nt < 2; ++nt) {
        v8f acc; for (int i = 0; i < 8; ++i) acc[i] = 0.f;
        acc = __builtin_amdgcn_wmma_f32_16x16x32_bf16(false, aq[mt][0].v, false, bk[nt][0].v,
                                                      (short)0, acc, false, false);
        acc = __builtin_amdgcn_wmma_f32_16x16x32_bf16(false, aq[mt][1].v, false, bk[nt][1].v,
                                                      (short)0, acc, false, false);
        const int kcol = k0 + nt * 16 + nn;
        if (isdiag) {
          for (int r = 0; r < 8; ++r)
            if (kcol > q0 + mt * 16 + 8 * hh + r) acc[r] = -1.0e9f;  // causal mask
        }
        for (int r = 0; r < 8; ++r) rmax[mt][r] = fmaxf(rmax[mt][r], acc[r]);
        v8bf pb = __builtin_convertvector(acc, v8bf);
        for (int r = 0; r < 8; ++r)
          Ps[(mt * 16 + 8 * hh + r) * PSTRIDE + kcol] = pb[r];  // transpose via LDS
      }
  }
  // reduce row-max across the 16 lanes of each half (halves own distinct rows)
  for (int mt = 0; mt < 2; ++mt)
    for (int r = 0; r < 8; ++r) {
      float v = rmax[mt][r];
      v = fmaxf(v, __shfl_xor(v, 1, 32));
      v = fmaxf(v, __shfl_xor(v, 2, 32));
      v = fmaxf(v, __shfl_xor(v, 4, 32));
      v = fmaxf(v, __shfl_xor(v, 8, 32));
      rmax[mt][r] = v;
    }
  if (nn == 0)
    for (int mt = 0; mt < 2; ++mt)
      for (int r = 0; r < 8; ++r)
        redbuf[wave * QTILE + mt * 16 + 8 * hh + r] = rmax[mt][r];
  __syncthreads();
  if (tid < QTILE) {
    float v = -3.0e38f;
    for (int w = 0; w < NWAVES; ++w) v = fmaxf(v, redbuf[w * QTILE + tid]);
    rowmax_s[tid] = v;
  }
  for (int idx = tid; idx < OACC_N; idx += 256) Oacc[idx] = 0.f;
  __syncthreads();

  // ==== Phase C: exp on A-frags (in place) + row sums + O_partial = P @ V ====
  float rmA[2] = { rowmax_s[nn], rowmax_s[16 + nn] };  // A layout: lane row = mt*16+nn
  float sumA[2] = { 0.f, 0.f };
  v8f accO[2][4];
  for (int mt = 0; mt < 2; ++mt)
    for (int nt = 0; nt < 4; ++nt)
      for (int i = 0; i < 8; ++i) accO[mt][nt][i] = 0.f;

#pragma unroll 2
  for (int c = wave; c < nchunks; c += NWAVES) {
    const int k0 = c * 32;
    if (c + NWAVES < nchunks)
      __builtin_prefetch(&Vbh[(size_t)(k0 + NWAVES * 32) * DK], 0, 0);

    FragBf ap[2];
    for (int mt = 0; mt < 2; ++mt) {
      const int base = (mt * 16 + nn) * PSTRIDE + k0 + 8 * hh;
      FragBf raw;
      raw.h[0] = *(const v8bf*)&Ps[base];
      raw.h[1] = *(const v8bf*)&Ps[base + 16];
      v8f rf0 = __builtin_convertvector(raw.h[0], v8f);
      v8f rf1 = __builtin_convertvector(raw.h[1], v8f);
      v8f e0, e1;
      for (int j = 0; j < 8; ++j) e0[j] = __expf(rf0[j] - rmA[mt]);
      for (int j = 0; j < 8; ++j) e1[j] = __expf(rf1[j] - rmA[mt]);
      for (int j = 0; j < 8; ++j) sumA[mt] += e0[j] + e1[j];
      ap[mt].h[0] = __builtin_convertvector(e0, v8bf);
      ap[mt].h[1] = __builtin_convertvector(e1, v8bf);
      *(v8bf*)&Ps[base]      = ap[mt].h[0];   // store exp'd probs for attn writeout
      *(v8bf*)&Ps[base + 16] = ap[mt].h[1];
    }
    for (int nt = 0; nt < 4; ++nt) {
      FragBf bv;   // lane: d-column nt*16+nn, key rows k0 + 8h + j (j<8), k0 + 16 + 8h + (j-8) (j>=8)
      v16f vv;
      for (int j = 0; j < 16; ++j)
        vv[j] = Vbh[(size_t)(k0 + 8 * hh + j + (j >= 8 ? 8 : 0)) * DK + nt * 16 + nn];
      bv.v = __builtin_convertvector(vv, v16bf);
      for (int mt = 0; mt < 2; ++mt)
        accO[mt][nt] = __builtin_amdgcn_wmma_f32_16x16x32_bf16(
            false, ap[mt].v, false, bv.v, (short)0, accO[mt][nt], false, false);
    }
  }
  // reduce row sums: lanes nn and nn+16 hold the same row (different k halves)
  for (int mt = 0; mt < 2; ++mt) sumA[mt] += __shfl_xor(sumA[mt], 16, 32);
  if (ln < 16) {
    redbuf[wave * QTILE + nn]      = sumA[0];
    redbuf[wave * QTILE + 16 + nn] = sumA[1];
  }
  __syncthreads();
  if (tid < QTILE) {
    float v = 0.f;
    for (int w = 0; w < NWAVES; ++w) v += redbuf[w * QTILE + tid];
    rowinv_s[tid] = 1.0f / v;
  }
  __syncthreads();

  // deterministic cross-wave O reduction, fragment-order LDS (conflict-free)
  for (int w = 0; w < NWAVES; ++w) {
    if (wave == w) {
      for (int mt = 0; mt < 2; ++mt)
        for (int nt = 0; nt < 4; ++nt)
          for (int r = 0; r < 8; ++r)
            Oacc[((mt * 4 + nt) * 8 + r) * 32 + ln] += accO[mt][nt][r];
    }
    __syncthreads();
  }

  // ---- write O = Oacc / l (invert fragment-order mapping) ----
  for (int idx = tid; idx < QTILE * DK; idx += 256) {
    const int row = idx >> 6, col = idx & 63;
    const int tile = (row >> 4) * 4 + (col >> 4);
    const int lane = ((row >> 3) & 1) * 16 + (col & 15);
    out[((size_t)bh * S_LEN + q0 + row) * DK + col] =
        Oacc[(tile * 8 + (row & 7)) * 32 + lane] * rowinv_s[row];
  }

  // ---- write normalized attn weights + causal zero fill (8 cols/iter) ----
  const int kmax = nchunks * 32;
  for (int idx = tid; idx < QTILE * (S_LEN / 8); idx += 256) {
    const int r   = idx >> 9;
    const int col = (idx & 511) * 8;
    v4f lo, hi;
    if (col < kmax) {
      const float inv = rowinv_s[r];
      const u32x4 u = *(const u32x4*)&Ps[r * PSTRIDE + col];   // 8 bf16, one b128
      union { uint32_t u; float f; } t;
      for (int j = 0; j < 4; ++j) {
        t.u = u[j] << 16;         lo[j] = 0.f; hi[j] = 0.f;    // init then overwrite
      }
      t.u = u[0] << 16;          lo[0] = t.f * inv;
      t.u = u[0] & 0xFFFF0000u;  lo[1] = t.f * inv;
      t.u = u[1] << 16;          lo[2] = t.f * inv;
      t.u = u[1] & 0xFFFF0000u;  lo[3] = t.f * inv;
      t.u = u[2] << 16;          hi[0] = t.f * inv;
      t.u = u[2] & 0xFFFF0000u;  hi[1] = t.f * inv;
      t.u = u[3] << 16;          hi[2] = t.f * inv;
      t.u = u[3] & 0xFFFF0000u;  hi[3] = t.f * inv;
    } else {
      for (int j = 0; j < 4; ++j) { lo[j] = 0.f; hi[j] = 0.f; }
    }
    float* dst = &attn[((size_t)bh * S_LEN + q0 + r) * S_LEN + col];
    *(v4f*)dst       = lo;
    *(v4f*)(dst + 4) = hi;
  }
}

extern "C" void kernel_launch(void* const* d_in, const int* in_sizes, int n_in,
                              void* d_out, int out_size, void* d_ws, size_t ws_size,
                              hipStream_t stream) {
  (void)in_sizes; (void)n_in; (void)out_size; (void)d_ws; (void)ws_size;
  const float* Q = (const float*)d_in[0];
  const float* K = (const float*)d_in[1];
  const float* V = (const float*)d_in[2];
  float* out  = (float*)d_out;
  float* attn = out + (size_t)2 * 8 * S_LEN * DK;   // outputs then attn weights, flat

  hipFuncSetAttribute((const void*)attn_fused_kernel,
                      hipFuncAttributeMaxDynamicSharedMemorySize, LDS_BYTES);

  dim3 grid(S_LEN / QTILE, 2 * 8);
  attn_fused_kernel<<<grid, dim3(256), LDS_BYTES, stream>>>(Q, K, V, out, attn);
}
